// AdaptiveMixing_65850438582899
// MI455X (gfx1250) — matrix operand order repaired and checked by reading, not compiled, not run on hardware
//
#include <hip/hip_runtime.h>
#include <hip/hip_bf16.h>
#include <math.h>

// ---------------------------------------------------------------------------
// AdaptiveMixing fused pipeline for MI455X (gfx1250), bf16 WMMA everywhere.
// B=4, T=2048 -> BT=8192 tokens. P=32 points, C=128 channels,
// QUERY_DIM=256, HIDDEN=64, TOTAL_PARAMS=17408, OUT=256.
//
// B operands are staged in LDS in "fragment panel" order so the WMMA inner
// loops read both A and B with 16-byte ds_load_b128 (no u16 gathers).
// Row-major A tiles are staged with the Tensor Data Mover (tensor_load_to_lds,
// TENSORcnt), overlapping the DMA with the B-panel swizzle.
// ---------------------------------------------------------------------------

#define BT        8192
#define QDIM      256
#define HID       64
#define CDIM      128
#define PPTS      32
#define TPAR      17408      // 128*128 + 32*32
#define CMSZ      16384
#define ODIM      256
#define FEAT      4096       // 32*128
#define CHUNK     512
#define NCHUNK    (BT / CHUNK)
#define PANEL     512        // elements per 32x16 B panel (32 lanes * 16)

typedef __attribute__((ext_vector_type(16))) __bf16 v16bf;
typedef __attribute__((ext_vector_type(8)))  __bf16 v8bf;
typedef __attribute__((ext_vector_type(8)))  float  v8f;
typedef __attribute__((ext_vector_type(4)))  unsigned int u32x4;
typedef __attribute__((ext_vector_type(8)))  int i32x8;
typedef __attribute__((ext_vector_type(4)))  int i32x4;

#define WMMA_BF16(A, B, C) \
  __builtin_amdgcn_wmma_f32_16x16x32_bf16(false, (A), false, (B), (short)0, (C), false, false)

__device__ __forceinline__ float gelu_exact(float x) {
  return 0.5f * x * (1.0f + erff(x * 0.70710678118654752f));
}

// Raw LDS byte offset of a shared-memory pointer (generic -> addrspace(3)).
__device__ __forceinline__ unsigned lds_off_u32(const void* p) {
  typedef __attribute__((address_space(3))) char lds_char;
  return (unsigned)(unsigned long long)(lds_char*)p;
}

// Tensor Data Mover: DMA a 2D tile of bf16 (tile_x elems per row, tile_y rows,
// row stride stride_x elems) from global into LDS at lds_off. Issued once per
// wave; completion tracked with TENSORcnt. D# per CDNA5 ISA ch.8.
// This toolchain's builtin takes 6 args (4 descriptor groups + pad + cpol).
__device__ __forceinline__ void tdm_load_2d_bf16(unsigned lds_off, const void* gaddr,
                                                 unsigned tile_x, unsigned tile_y,
                                                 unsigned tensor_x, unsigned tensor_y,
                                                 unsigned stride_x) {
  const unsigned long long ga = (unsigned long long)(size_t)gaddr;
  u32x4 g0;
  g0[0] = 1u;                                            // count=1 (valid), user mode
  g0[1] = lds_off;                                       // lds_addr (bytes)
  g0[2] = (unsigned)(ga & 0xffffffffu);                  // global_addr[31:0]
  g0[3] = (unsigned)((ga >> 32) & 0x01ffffffu) | (2u << 30);  // addr[56:32] | type=2
  i32x8 g1;
  g1[0] = (int)(1u << 16);                               // data_size=1 (2 bytes)
  g1[1] = (int)((tensor_x & 0xffffu) << 16);             // tensor_dim0[15:0]
  g1[2] = (int)((tensor_x >> 16) | ((tensor_y & 0xffffu) << 16));
  g1[3] = (int)((tensor_y >> 16) | (tile_x << 16));      // tile_dim0
  g1[4] = (int)(tile_y & 0xffffu);                       // tile_dim1 (tile_dim2=0)
  g1[5] = (int)stride_x;                                 // tensor_dim0_stride[31:0]
  g1[6] = 0;
  g1[7] = 0;
  const i32x4 z4 = {0, 0, 0, 0};                         // 2D: groups 2/3 unused
  const i32x8 z8 = {0, 0, 0, 0, 0, 0, 0, 0};
  __builtin_amdgcn_tensor_load_to_lds(g0, g1, z4, z4, z8, 0);
}

// A fragment: 16x32 bf16 tile, row-major in LDS (row stride ld, ld % 8 == 0).
// Per lane the data is two contiguous 16-byte runs -> two ds_load_b128.
__device__ __forceinline__ v16bf load_a_frag(const __bf16* __restrict__ src, int ld) {
  const int lane = threadIdx.x & 31;
  const int l = lane & 15, hi = lane >> 4;
  const __bf16* row = src + l * ld + hi * 16;
  v8bf lo = *(const v8bf*)(row);       // k = hi*16 .. hi*16+7
  v8bf hv = *(const v8bf*)(row + 8);   // k = hi*16+8 .. hi*16+15
  return __builtin_shufflevector(lo, hv, 0, 1, 2, 3, 4, 5, 6, 7,
                                 8, 9, 10, 11, 12, 13, 14, 15);
}

// B fragment from a pre-swizzled panel: per lane 16 contiguous bf16 (32 B).
__device__ __forceinline__ v16bf load_b_panel(const __bf16* __restrict__ panel) {
  return *(const v16bf*)(panel + (threadIdx.x & 31) * 16);
}

// LDS offset for element (k, n) of a B tile staged as panels.
// ntiles = (#columns)/16. panel = (k>>5)*ntiles + (n>>4);
// within panel: lane = (k>>4 & 1)*16 + (n&15), element = k & 15.
__device__ __forceinline__ int bpanel_idx(int k, int n, int ntiles) {
  return ((k >> 5) * ntiles + (n >> 4)) * PANEL +
         (((k >> 4) & 1) * 16 + (n & 15)) * 16 + (k & 15);
}

// ---------------------------------------------------------------------------
// Kernel 0: one-time weight conversion. w2 -> bf16 (same layout),
// wo[256 x 4096] -> WOB[4096 x 256] bf16 (transposed so it is the GEMM B).
// ---------------------------------------------------------------------------
__global__ void convert_weights_kernel(const float* __restrict__ w2,
                                       const float* __restrict__ wo,
                                       unsigned short* __restrict__ w2b_u16,
                                       unsigned short* __restrict__ wob_u16) {
  __bf16* W2B = (__bf16*)w2b_u16;
  __bf16* WOB = (__bf16*)wob_u16;
  const int stride = gridDim.x * blockDim.x;
  const int i0 = blockIdx.x * blockDim.x + threadIdx.x;
  for (int i = i0; i < TPAR * HID; i += stride) W2B[i] = (__bf16)w2[i];
  for (int i = i0; i < FEAT * ODIM; i += stride) {
    const int n = i & (ODIM - 1);
    const int k = i >> 8;
    WOB[i] = (__bf16)wo[n * FEAT + k];   // WOB[k][n] = wo[n][k]
  }
}

// ---------------------------------------------------------------------------
// Kernel 1: per token LayerNorm(query) then h = qn @ w1^T + b1.
// One wave per token (wave32). Stores H transposed: HT[k][t] bf16 (k<64).
// ---------------------------------------------------------------------------
__global__ void ln_h_kernel(const float* __restrict__ query,
                            const float* __restrict__ ln_w,
                            const float* __restrict__ ln_b,
                            const float* __restrict__ w1,
                            const float* __restrict__ b1,
                            unsigned short* __restrict__ ht_u16) {
  __bf16* HT = (__bf16*)ht_u16;
  __shared__ float qn[8][QDIM];
  const int wave = threadIdx.x >> 5;
  const int lane = threadIdx.x & 31;
  const int t = blockIdx.x * 8 + wave;

  const float* q = query + (size_t)t * QDIM;
  float v[8], s = 0.f, ss = 0.f;
#pragma unroll
  for (int j = 0; j < 8; ++j) {
    v[j] = q[lane + 32 * j];
    s += v[j];
    ss += v[j] * v[j];
  }
#pragma unroll
  for (int off = 16; off > 0; off >>= 1) {
    s += __shfl_xor(s, off, 32);
    ss += __shfl_xor(ss, off, 32);
  }
  const float mu  = s * (1.0f / QDIM);
  const float var = ss * (1.0f / QDIM) - mu * mu;
  const float rs  = rsqrtf(var + 1e-6f);
#pragma unroll
  for (int j = 0; j < 8; ++j) {
    const int i = lane + 32 * j;
    qn[wave][i] = (v[j] - mu) * rs * ln_w[i] + ln_b[i];
  }
  __syncthreads();
#pragma unroll
  for (int r = 0; r < 2; ++r) {
    const int jj = lane + 32 * r;          // 0..63
    float acc = b1[jj];
    const float* wr = w1 + jj * QDIM;
    for (int i = 0; i < QDIM; ++i) acc += qn[wave][i] * wr[i];
    HT[jj * BT + t] = (__bf16)acc;
  }
}

// ---------------------------------------------------------------------------
// Kernel 2: paramgen GEMM  P[17408 x 512chunk] = W2B[17408x64] @ HT[64xchunk] + b2
// stored per-token contiguous: PB[token_local][m] bf16.
// WG tile 128M x 128N, 8 waves in 4x2 grid (wave tile 32M x 64N).
// A tile comes in via TDM; epilogue goes through LDS for coalesced stores.
// ---------------------------------------------------------------------------
__global__ void paramgen_kernel(const unsigned short* __restrict__ w2b_u16,
                                const unsigned short* __restrict__ ht_u16,
                                const float* __restrict__ b2,
                                unsigned short* __restrict__ pb_u16,
                                int chunk0) {
  const __bf16* W2B = (const __bf16*)w2b_u16;
  const __bf16* HT  = (const __bf16*)ht_u16;
  __bf16* PB        = (__bf16*)pb_u16;

  __shared__ __align__(32) __bf16 As[128 * HID];   // 16 KB, row-major
  __shared__ __align__(32) __bf16 Bs[HID * 128];   // 16 KB, panels (2 kb x 8 nt)
  __shared__ __align__(32) __bf16 Cs[128 * 128];   // 32 KB, [n_local][m_local]
  __shared__ float b2s[128];

  const int m0 = blockIdx.x * 128;   // over 17408 = 136 tiles
  const int n0 = blockIdx.y * 128;   // over CHUNK = 4 tiles
  const int tid = threadIdx.x;

  // A: rows m0..m0+127, k 0..63 via Tensor Data Mover (wave 0 issues).
  if (tid < 32) {
    tdm_load_2d_bf16(lds_off_u32(As), W2B + (size_t)m0 * HID,
                     /*tile_x=*/HID, /*tile_y=*/128,
                     /*tensor_x=*/HID, /*tensor_y=*/TPAR, /*stride_x=*/HID);
  }
  // B: Bs panels from HT[k][chunk0+n0+n] (overlaps with the TDM).
  for (int idx = tid; idx < 1024; idx += 256) {
    const int k = idx >> 4, n8 = (idx & 15) * 8;
    uint4 raw = *(const uint4*)(HT + (size_t)k * BT + chunk0 + n0 + n8);
    const __bf16* v = (const __bf16*)&raw;
#pragma unroll
    for (int j = 0; j < 8; ++j) Bs[bpanel_idx(k, n8 + j, 8)] = v[j];
  }
  if (tid < 128) b2s[tid] = b2[m0 + tid];
  if (tid < 32) __builtin_amdgcn_s_wait_tensorcnt(0);
  __syncthreads();

  const int wave = tid >> 5, lane = tid & 31;
  const int wm = wave >> 1, wn = wave & 1;
  const v8f vzero = {0.f, 0.f, 0.f, 0.f, 0.f, 0.f, 0.f, 0.f};
  v8f acc[2][4];
#pragma unroll
  for (int mt = 0; mt < 2; ++mt)
#pragma unroll
    for (int nt = 0; nt < 4; ++nt) acc[mt][nt] = vzero;

#pragma unroll
  for (int k0 = 0; k0 < HID; k0 += 32) {
    v16bf a0 = load_a_frag(As + (wm * 32 + 0) * HID + k0, HID);
    v16bf a1 = load_a_frag(As + (wm * 32 + 16) * HID + k0, HID);
#pragma unroll
    for (int nt = 0; nt < 4; ++nt) {
      v16bf b = load_b_panel(Bs + ((k0 >> 5) * 8 + wn * 4 + nt) * PANEL);
      acc[0][nt] = WMMA_BF16(a0, b, acc[0][nt]);
      acc[1][nt] = WMMA_BF16(a1, b, acc[1][nt]);
    }
  }

  const int l = lane & 15, hi = lane >> 4;
#pragma unroll
  for (int mt = 0; mt < 2; ++mt)
#pragma unroll
    for (int nt = 0; nt < 4; ++nt) {
#pragma unroll
      for (int v = 0; v < 8; ++v) {
        const int ml = wm * 32 + mt * 16 + v + 8 * hi;       // m within tile
        const int nl = wn * 64 + nt * 16 + l;                // n within tile
        Cs[nl * 128 + ml] = (__bf16)(acc[mt][nt][v] + b2s[ml]);
      }
    }
  __syncthreads();

  // Coalesced store: each row n -> 256 contiguous bytes of PB.
  for (int idx = tid; idx < 2048; idx += 256) {
    const int n = idx >> 4, q = idx & 15;
    ((uint4*)(PB + (size_t)(n0 + n) * TPAR + m0))[q] =
        ((const uint4*)(Cs + n * 128))[q];
  }
}

// ---------------------------------------------------------------------------
// Kernel 3: per-token mixing. One WG per token, 8 waves.
//  Y1 = gelu(X[32x128] @ CM[128x128] + m_beta)
//  Y2 = gelu(SM[32x32] @ Y1 + s_beta)     -> Y2B[token_local][4096] bf16
// Wave w owns 16 output columns (d = w*16 .. w*16+15), M rows 0..31.
// CM and Y1 are kept in B-panel order; Y1 panels are written directly by the
// channel-mix epilogue.
// ---------------------------------------------------------------------------
__global__ void mixing_kernel(const float* __restrict__ sampled,
                              const unsigned short* __restrict__ pb_u16,
                              const float* __restrict__ m_beta,
                              const float* __restrict__ s_beta,
                              unsigned short* __restrict__ y2b_u16,
                              int chunk0) {
  const __bf16* PB = (const __bf16*)pb_u16;
  __bf16* Y2B      = (__bf16*)y2b_u16;

  __shared__ __align__(32) __bf16 Xs[PPTS * CDIM];    //  8 KB row-major (reused for Y2)
  __shared__ __align__(32) __bf16 CMs[CDIM * CDIM];   // 32 KB panels (4 kb x 8 nt)
  __shared__ __align__(32) __bf16 SMs[PPTS * PPTS];   //  2 KB row-major (A matrix)
  __shared__ __align__(32) __bf16 Y1s[PPTS * CDIM];   //  8 KB panels (1 kb x 8 nt)

  const int tl = blockIdx.x;                       // token within chunk
  const size_t t = (size_t)chunk0 + tl;
  const int tid = threadIdx.x;

  // Stage X (f32 -> bf16, row-major).
  const float* X = sampled + t * (PPTS * CDIM);
  for (int idx = tid; idx < 1024; idx += 256) {
    float4 f = ((const float4*)X)[idx];
    __bf16* d = Xs + idx * 4;
    d[0] = (__bf16)f.x; d[1] = (__bf16)f.y;
    d[2] = (__bf16)f.z; d[3] = (__bf16)f.w;
  }
  // Stage CM into panels.
  const __bf16* P = PB + (size_t)tl * TPAR;
  for (int idx = tid; idx < 2048; idx += 256) {
    const int c = (idx * 8) >> 7, d0 = (idx * 8) & 127;
    uint4 raw = *(const uint4*)(P + idx * 8);
    const __bf16* v = (const __bf16*)&raw;
#pragma unroll
    for (int j = 0; j < 8; ++j) CMs[bpanel_idx(c, d0 + j, 8)] = v[j];
  }
  // Stage SM row-major (it is the A matrix of the spatial GEMM).
  for (int idx = tid; idx < 128; idx += 256)
    *(uint4*)(SMs + idx * 8) = *(const uint4*)(P + CMSZ + idx * 8);
  __syncthreads();

  const int wave = tid >> 5, lane = tid & 31;
  const int l = lane & 15, hi = lane >> 4;
  const int ncol = wave * 16;
  const v8f vzero = {0.f, 0.f, 0.f, 0.f, 0.f, 0.f, 0.f, 0.f};

  // ---- channel mixing: M=32, N=16 (this wave), K=128 ----
  v8f c0 = vzero, c1 = vzero;
#pragma unroll
  for (int k0 = 0; k0 < CDIM; k0 += 32) {
    v16bf a0 = load_a_frag(Xs + 0 * CDIM + k0, CDIM);
    v16bf a1 = load_a_frag(Xs + 16 * CDIM + k0, CDIM);
    v16bf b  = load_b_panel(CMs + ((k0 >> 5) * 8 + wave) * PANEL);
    c0 = WMMA_BF16(a0, b, c0);
    c1 = WMMA_BF16(a1, b, c1);
  }
  // Epilogue straight into Y1 panels: dest = wave*PANEL + (mt*16+l)*16 + v + 8*hi.
  const float mb = m_beta[ncol + l];
  __bf16* yp = Y1s + wave * PANEL + l * 16 + 8 * hi;
#pragma unroll
  for (int v = 0; v < 8; ++v) {
    yp[v]       = (__bf16)gelu_exact(c0[v] + mb);   // mt = 0 (p = v+8hi)
    yp[256 + v] = (__bf16)gelu_exact(c1[v] + mb);   // mt = 1 (p = 16+v+8hi)
  }
  __syncthreads();

  // ---- spatial mixing: A = SM[32x32], B = Y1[32x128] (panels), K=32 ----
  v16bf sa0 = load_a_frag(SMs + 0 * PPTS, PPTS);
  v16bf sa1 = load_a_frag(SMs + 16 * PPTS, PPTS);
  v16bf sb  = load_b_panel(Y1s + wave * PANEL);
  v8f o0 = vzero, o1 = vzero;
  o0 = WMMA_BF16(sa0, sb, o0);
  o1 = WMMA_BF16(sa1, sb, o1);

  // Stage Y2 into LDS (reuse Xs; it is dead after the channel stage),
  // then do 16-byte coalesced global stores.
#pragma unroll
  for (int v = 0; v < 8; ++v) {
    const int oa = v + 8 * hi;
    const int ob = 16 + v + 8 * hi;
    Xs[oa * CDIM + ncol + l] = (__bf16)gelu_exact(o0[v] + s_beta[oa]);
    Xs[ob * CDIM + ncol + l] = (__bf16)gelu_exact(o1[v] + s_beta[ob]);
  }
  __syncthreads();
  uint4* dst = (uint4*)(Y2B + (size_t)tl * FEAT);
  const uint4* src = (const uint4*)Xs;
  for (int idx = tid; idx < 512; idx += 256) dst[idx] = src[idx];
}

// ---------------------------------------------------------------------------
// Kernel 4: out-proj GEMM  out[chunk x 256] = Y2B[chunk x 4096] @ WOB[4096 x 256] + bo
// WG tile 128M x 128N, 8 waves 4x2. Per K step the A tile is DMA'd by the TDM
// while all threads swizzle the B tile into panels.
// ---------------------------------------------------------------------------
__global__ void outproj_kernel(const unsigned short* __restrict__ y2b_u16,
                               const unsigned short* __restrict__ wob_u16,
                               const float* __restrict__ bo,
                               float* __restrict__ out,
                               int chunk0) {
  const __bf16* Y2B = (const __bf16*)y2b_u16;
  const __bf16* WOB = (const __bf16*)wob_u16;

  __shared__ __align__(32) __bf16 As[128 * 32];   // 8 KB row-major
  __shared__ __align__(32) __bf16 Bs[32 * 128];   // 8 KB panels (1 kb x 8 nt)

  const int m0 = blockIdx.x * 128;  // tokens within chunk (4 tiles)
  const int n0 = blockIdx.y * 128;  // output dim (2 tiles)
  const int tid = threadIdx.x;
  const int wave = tid >> 5, lane = tid & 31;
  const int wm = wave >> 1, wn = wave & 1;
  const v8f vzero = {0.f, 0.f, 0.f, 0.f, 0.f, 0.f, 0.f, 0.f};
  v8f acc[2][4];
#pragma unroll
  for (int mt = 0; mt < 2; ++mt)
#pragma unroll
    for (int nt = 0; nt < 4; ++nt) acc[mt][nt] = vzero;

  const unsigned as_off = lds_off_u32(As);

  for (int k0 = 0; k0 < FEAT; k0 += 32) {
    __syncthreads();
    // A: [128 tokens x 32 k] via TDM (wave 0), overlapped with B swizzle.
    if (tid < 32) {
      tdm_load_2d_bf16(as_off, Y2B + (size_t)m0 * FEAT + k0,
                       /*tile_x=*/32, /*tile_y=*/128,
                       /*tensor_x=*/FEAT, /*tensor_y=*/CHUNK, /*stride_x=*/FEAT);
    }
    // B: [32 k x 128 n] -> panels.
    for (int idx = tid; idx < 512; idx += 256) {
      const int k = idx >> 4, n8 = (idx & 15) * 8;
      uint4 raw = *(const uint4*)(WOB + (size_t)(k0 + k) * ODIM + n0 + n8);
      const __bf16* v = (const __bf16*)&raw;
#pragma unroll
      for (int j = 0; j < 8; ++j) Bs[bpanel_idx(k, n8 + j, 8)] = v[j];
    }
    // Prefetch next K tile of B into cache (global_prefetch_b8).
    if (k0 + 32 < FEAT)
      __builtin_prefetch(WOB + (size_t)(k0 + 32) * ODIM + n0, 0, 1);
    if (tid < 32) __builtin_amdgcn_s_wait_tensorcnt(0);
    __syncthreads();

    v16bf a0 = load_a_frag(As + (wm * 32 + 0) * 32, 32);
    v16bf a1 = load_a_frag(As + (wm * 32 + 16) * 32, 32);
#pragma unroll
    for (int nt = 0; nt < 4; ++nt) {
      v16bf b = load_b_panel(Bs + (wn * 4 + nt) * PANEL);
      acc[0][nt] = WMMA_BF16(a0, b, acc[0][nt]);
      acc[1][nt] = WMMA_BF16(a1, b, acc[1][nt]);
    }
  }

  const int l = lane & 15, hi = lane >> 4;
#pragma unroll
  for (int mt = 0; mt < 2; ++mt)
#pragma unroll
    for (int nt = 0; nt < 4; ++nt) {
#pragma unroll
      for (int v = 0; v < 8; ++v) {
        const int m = m0 + wm * 32 + mt * 16 + v + 8 * hi;   // token within chunk
        const int n = n0 + wn * 64 + nt * 16 + l;
        out[(size_t)(chunk0 + m) * ODIM + n] = acc[mt][nt][v] + bo[n];
      }
    }
}

// ---------------------------------------------------------------------------
// Host launcher
// ---------------------------------------------------------------------------
extern "C" void kernel_launch(void* const* d_in, const int* in_sizes, int n_in,
                              void* d_out, int out_size, void* d_ws, size_t ws_size,
                              hipStream_t stream) {
  const float* sampled = (const float*)d_in[0];
  const float* query   = (const float*)d_in[1];
  const float* ln_w    = (const float*)d_in[2];
  const float* ln_b    = (const float*)d_in[3];
  const float* w1      = (const float*)d_in[4];
  const float* b1      = (const float*)d_in[5];
  const float* w2      = (const float*)d_in[6];
  const float* b2      = (const float*)d_in[7];
  const float* m_beta  = (const float*)d_in[8];
  const float* s_beta  = (const float*)d_in[9];
  const float* wo      = (const float*)d_in[10];
  const float* bo      = (const float*)d_in[11];
  float* out = (float*)d_out;

  // workspace carve-up (bf16 buffers, 256B aligned)
  char* ws = (char*)d_ws;
  size_t off = 0;
  auto carve = [&](size_t bytes) {
    void* p = ws + off;
    off += (bytes + 255) & ~(size_t)255;
    return p;
  };
  unsigned short* W2B = (unsigned short*)carve((size_t)TPAR * HID * 2);   // 2.2 MB
  unsigned short* HT  = (unsigned short*)carve((size_t)HID * BT * 2);     // 1.0 MB
  unsigned short* WOB = (unsigned short*)carve((size_t)FEAT * ODIM * 2);  // 2.0 MB
  unsigned short* PB  = (unsigned short*)carve((size_t)CHUNK * TPAR * 2); // 17.8 MB
  unsigned short* Y2B = (unsigned short*)carve((size_t)CHUNK * FEAT * 2); // 4.0 MB
  (void)ws_size; (void)in_sizes; (void)n_in; (void)out_size;

  convert_weights_kernel<<<2048, 256, 0, stream>>>(w2, wo, W2B, WOB);
  ln_h_kernel<<<BT / 8, 256, 0, stream>>>(query, ln_w, ln_b, w1, b1, HT);

  for (int c = 0; c < NCHUNK; ++c) {
    const int chunk0 = c * CHUNK;
    paramgen_kernel<<<dim3(TPAR / 128, CHUNK / 128), 256, 0, stream>>>(
        W2B, HT, b2, PB, chunk0);
    mixing_kernel<<<CHUNK, 256, 0, stream>>>(
        sampled, PB, m_beta, s_beta, Y2B, chunk0);
    outproj_kernel<<<dim3(CHUNK / 128, ODIM / 128), 256, 0, stream>>>(
        Y2B, WOB, bo, out, chunk0);
  }
}
